// GIN_10892037062711
// MI455X (gfx1250) — matrix operand order repaired and checked by reading, not compiled, not run on hardware
//
#include <hip/hip_runtime.h>

typedef __attribute__((ext_vector_type(16))) _Float16 v16h;
typedef __attribute__((ext_vector_type(8)))  float    v8f;

#define HID 64
#define BN_EPS 1e-5f
#define LDSPAD 72   // padded LDS row stride (halfs) to dodge bank conflicts

// ---------------------------------------------------------------------------
// Convert 4 64x64 fp32 weight matrices -> f16, pre-shuffled into WMMA
// B-fragment order so each wave can load its fragment as one aligned v16h:
//   dst[m*4096 + nt*1024 + s*512 + lane*16 + e] = W_m[k*64 + n]
//   where half = lane>>4, l16 = lane&15, k = s*32 + 16*half + e, n = nt*16+l16
// ---------------------------------------------------------------------------
__global__ __launch_bounds__(256) void gin_cvtw_kernel(
    const float* __restrict__ a, const float* __restrict__ b,
    const float* __restrict__ c, const float* __restrict__ d,
    _Float16* __restrict__ dst)
{
    int t = blockIdx.x * 256 + threadIdx.x;       // < 4*4096
    int m    = t >> 12;
    int r    = t & 4095;
    int e    = r & 15;
    int lane = (r >> 4) & 31;
    int s    = (r >> 9) & 1;
    int nt   = (r >> 10) & 3;
    int k    = s * 32 + 16 * (lane >> 4) + e;
    int n    = nt * 16 + (lane & 15);
    const float* w = (m == 0) ? a : (m == 1) ? b : (m == 2) ? c : d;
    dst[t] = (_Float16)w[k * HID + n];
}

// ---------------------------------------------------------------------------
// Edge scatter-add: agg[dst] += h[src], 64 lanes per edge (coalesced per row)
// ---------------------------------------------------------------------------
__global__ __launch_bounds__(256) void gin_scatter_kernel(
    const float* __restrict__ h, const long long* __restrict__ src,
    const long long* __restrict__ dst, float* __restrict__ agg, int nEdges)
{
    int e = blockIdx.x * 4 + (threadIdx.x >> 6);
    int f = threadIdx.x & 63;
    if (e < nEdges) {
        int s = (int)src[e];
        int d = (int)dst[e];
        atomicAdd(&agg[d * HID + f], h[s * HID + f]);
    }
}

// ---------------------------------------------------------------------------
// Fused GEMM (N x 64) @ (64 x 64) with WMMA f16->f32.
//   mode 0: z = in + in2                      (GIN aggregation: h + agg)
//   mode 1: z = relu(in*scale[c] + shift[c])  (BN+ReLU prologue)
// Weights arrive pre-shuffled in B-fragment order (see gin_cvtw_kernel).
// Epilogue: out = tile + bias[col]; accumulates per-feature sum/sumsq for BN.
// Requires N % 32 == 0 (100000 = 3125*32) so EXEC is all-ones for WMMA.
// ---------------------------------------------------------------------------
__global__ __launch_bounds__(256) void gin_gemm_kernel(
    const float* __restrict__ in, const float* __restrict__ in2,
    const float* __restrict__ scale, const float* __restrict__ shift,
    const _Float16* __restrict__ w, const float* __restrict__ bias,
    float* __restrict__ out, float* __restrict__ sum, float* __restrict__ sumsq,
    int mode)
{
    __shared__ _Float16 zt[32][LDSPAD];
    __shared__ float ssum[HID], ssq[HID];

    int tid = threadIdx.x;
    int m0  = blockIdx.x * 32;

    if (tid < HID) { ssum[tid] = 0.f; ssq[tid] = 0.f; }

    // Stage 32x64 activation tile into LDS as f16 (8 floats per thread).
    int r  = tid >> 3;
    int c0 = (tid & 7) * 8;
    int base = (m0 + r) * HID + c0;
    if (mode == 0) {
        #pragma unroll
        for (int j = 0; j < 8; ++j)
            zt[r][c0 + j] = (_Float16)(in[base + j] + in2[base + j]);
    } else {
        #pragma unroll
        for (int j = 0; j < 8; ++j) {
            int cc = c0 + j;
            float z = fmaf(in[base + j], scale[cc], shift[cc]);
            zt[r][cc] = (_Float16)fmaxf(z, 0.f);
        }
    }
    __syncthreads();

    int wv   = tid >> 5;          // wave 0..7
    int lane = tid & 31;
    int half = lane >> 4;
    int l16  = lane & 15;
    int mt   = wv >> 2;           // 0..1 -> row tile
    int nt   = wv & 3;            // 0..3 -> col tile
    int mrow = mt * 16 + l16;
    int ncol = nt * 16 + l16;

    const v16h* wfrag = (const v16h*)w;   // 32B-aligned fragment chunks

    v8f acc = {};
    #pragma unroll
    for (int s = 0; s < 2; ++s) {
        int k0 = s * 32;
        v16h a;
        #pragma unroll
        for (int e = 0; e < 16; ++e) {
            // A (16x32 f16): VGPR e/2, K = e + 16*half (matches packed B)
            int ka = k0 + 16 * half + e;
            a[e] = zt[mrow][ka];
        }
        // B fragment: one aligned 32-byte load per lane (pre-shuffled layout)
        v16h b = wfrag[(nt * 2 + s) * 32 + lane];
        acc = __builtin_amdgcn_wmma_f32_16x16x32_f16(
            /*neg_a=*/false, a, /*neg_b=*/false, b,
            /*c_mod=*/(short)0, acc, /*reuse_a=*/false, /*reuse_b=*/false);
    }

    // Epilogue: bias add, store, per-feature BN statistics.
    float bcol = bias[ncol];
    float ps = 0.f, pq = 0.f;
    #pragma unroll
    for (int i = 0; i < 8; ++i) {
        int rl = mt * 16 + half * 8 + i;   // C layout: VGPR i -> row i + 8*half
        float v = acc[i] + bcol;
        out[(m0 + rl) * HID + ncol] = v;
        ps += v;
        pq += v * v;
    }
    atomicAdd(&ssum[ncol], ps);
    atomicAdd(&ssq[ncol], pq);
    __syncthreads();
    if (tid < HID) {
        atomicAdd(&sum[tid],   ssum[tid]);
        atomicAdd(&sumsq[tid], ssq[tid]);
    }
}

// ---------------------------------------------------------------------------
// Fold sum/sumsq -> per-feature scale/shift:  y = x*scale + shift
// ---------------------------------------------------------------------------
__global__ void gin_bnfin_kernel(
    const float* __restrict__ sum, const float* __restrict__ sumsq,
    const float* __restrict__ gamma, const float* __restrict__ beta,
    float* __restrict__ scale, float* __restrict__ shift, float invN)
{
    int f = threadIdx.x;
    if (f < HID) {
        float mean = sum[f] * invN;
        float var  = sumsq[f] * invN - mean * mean;
        float inv  = rsqrtf(var + BN_EPS);
        float sc   = gamma[f] * inv;
        scale[f] = sc;
        shift[f] = beta[f] - mean * sc;
    }
}

// ---------------------------------------------------------------------------
// Elementwise h = relu(x*scale + shift)
// ---------------------------------------------------------------------------
__global__ __launch_bounds__(256) void gin_bnrelu_kernel(
    const float* __restrict__ in, const float* __restrict__ scale,
    const float* __restrict__ shift, float* __restrict__ out, int n)
{
    int i = blockIdx.x * 256 + threadIdx.x;
    if (i < n) {
        int c = i & 63;
        out[i] = fmaxf(fmaf(in[i], scale[c], shift[c]), 0.f);
    }
}

// ---------------------------------------------------------------------------
// Global add-pool per graph: pooled[batch[node]] += h[node]
// ---------------------------------------------------------------------------
__global__ __launch_bounds__(256) void gin_pool_kernel(
    const float* __restrict__ h, const long long* __restrict__ batch,
    float* __restrict__ pooled, int n)
{
    int i = blockIdx.x * 256 + threadIdx.x;
    if (i < n) {
        int node = i >> 6;
        int f    = i & 63;
        int g    = (int)batch[node];
        atomicAdd(&pooled[g * HID + f], h[i]);
    }
}

// ---------------------------------------------------------------------------
// Readout: out[g,c] = sum_l pooled_l[g,:] @ wl[:,c] + bl[c]
// ---------------------------------------------------------------------------
__global__ __launch_bounds__(256) void gin_readout_kernel(
    const float* __restrict__ p0, const float* __restrict__ p1,
    const float* __restrict__ p2,
    const float* __restrict__ w0, const float* __restrict__ b0,
    const float* __restrict__ w1, const float* __restrict__ b1,
    const float* __restrict__ w2, const float* __restrict__ b2,
    float* __restrict__ out, int nG)
{
    int t = blockIdx.x * 256 + threadIdx.x;
    if (t < nG * 10) {
        int g = t / 10, c = t % 10;
        const float* r0 = p0 + g * HID;
        const float* r1 = p1 + g * HID;
        const float* r2 = p2 + g * HID;
        float acc = b0[c] + b1[c] + b2[c];
        #pragma unroll 8
        for (int f = 0; f < HID; ++f)
            acc += r0[f] * w0[f * 10 + c]
                 + r1[f] * w1[f * 10 + c]
                 + r2[f] * w2[f * 10 + c];
        out[t] = acc;
    }
}

// ---------------------------------------------------------------------------
extern "C" void kernel_launch(void* const* d_in, const int* in_sizes, int n_in,
                              void* d_out, int out_size, void* d_ws, size_t ws_size,
                              hipStream_t stream)
{
    const int N_ = in_sizes[0] / HID;       // 100000
    const int E_ = in_sizes[1] / 2;         // 1600000
    const int G_ = out_size / 10;           // 1000

    const float*     x     = (const float*)d_in[0];
    const long long* src   = (const long long*)d_in[1];
    const long long* dst   = src + E_;
    const long long* batch = (const long long*)d_in[2];
    const float* c0_w1 = (const float*)d_in[3];
    const float* c0_b1 = (const float*)d_in[4];
    const float* c0_g1 = (const float*)d_in[5];
    const float* c0_be1= (const float*)d_in[6];
    const float* c0_w2 = (const float*)d_in[7];
    const float* c0_b2 = (const float*)d_in[8];
    const float* bn0_g = (const float*)d_in[9];
    const float* bn0_b = (const float*)d_in[10];
    const float* c1_w1 = (const float*)d_in[11];
    const float* c1_b1 = (const float*)d_in[12];
    const float* c1_g1 = (const float*)d_in[13];
    const float* c1_be1= (const float*)d_in[14];
    const float* c1_w2 = (const float*)d_in[15];
    const float* c1_b2 = (const float*)d_in[16];
    const float* bn1_g = (const float*)d_in[17];
    const float* bn1_b = (const float*)d_in[18];
    const float* l0_w  = (const float*)d_in[19];
    const float* l0_b  = (const float*)d_in[20];
    const float* l1_w  = (const float*)d_in[21];
    const float* l1_b  = (const float*)d_in[22];
    const float* l2_w  = (const float*)d_in[23];
    const float* l2_b  = (const float*)d_in[24];
    float* out = (float*)d_out;

    // Workspace carve-up (256B aligned)
    char* ws = (char*)d_ws;
    size_t off = 0;
    auto carve = [&](size_t bytes) {
        void* p = ws + off;
        off += (bytes + 255) & ~(size_t)255;
        return p;
    };
    const size_t nhB = (size_t)N_ * HID * sizeof(float);
    float*    bufA   = (float*)carve(nhB);            // agg / y2 ping-pong
    float*    bufB   = (float*)carve(nhB);
    float*    h1     = (float*)carve(nhB);
    float*    h2     = (float*)carve(nhB);
    _Float16* wf16   = (_Float16*)carve(4 * 4096 * sizeof(_Float16));
    float*    stats  = (float*)carve(4 * 128 * sizeof(float));   // 4 BN x {sum,sumsq}
    float*    scsh   = (float*)carve(4 * 128 * sizeof(float));   // 4 BN x {scale,shift}
    float*    pooled = (float*)carve((size_t)3 * G_ * HID * sizeof(float));

    const float invN = 1.0f / (float)N_;
    const int gemmGrid = N_ / 32;                       // 3125 (N divisible by 32)
    const int ewGrid   = (N_ * HID + 255) / 256;        // elementwise / pool grids
    const int scGrid   = (E_ + 3) / 4;

    // Zero accumulators (re-done every call: deterministic, graph-safe)
    hipMemsetAsync(bufA,   0, nhB, stream);
    hipMemsetAsync(stats,  0, 4 * 128 * sizeof(float), stream);
    hipMemsetAsync(pooled, 0, (size_t)3 * G_ * HID * sizeof(float), stream);

    gin_cvtw_kernel<<<64, 256, 0, stream>>>(c0_w1, c0_w2, c1_w1, c1_w2, wf16);
    gin_pool_kernel<<<ewGrid, 256, 0, stream>>>(x, batch, pooled, N_ * HID);

    // ---- Layer 0 ----
    gin_scatter_kernel<<<scGrid, 256, 0, stream>>>(x, src, dst, bufA, E_);
    gin_gemm_kernel<<<gemmGrid, 256, 0, stream>>>(x, bufA, nullptr, nullptr,
        wf16 + 0 * 4096, c0_b1, bufB, stats + 0, stats + 64, 0);
    gin_bnfin_kernel<<<1, 64, 0, stream>>>(stats + 0, stats + 64, c0_g1, c0_be1,
        scsh + 0, scsh + 64, invN);
    gin_gemm_kernel<<<gemmGrid, 256, 0, stream>>>(bufB, nullptr, scsh + 0, scsh + 64,
        wf16 + 1 * 4096, c0_b2, bufA, stats + 128, stats + 192, 1);
    gin_bnfin_kernel<<<1, 64, 0, stream>>>(stats + 128, stats + 192, bn0_g, bn0_b,
        scsh + 128, scsh + 192, invN);
    gin_bnrelu_kernel<<<ewGrid, 256, 0, stream>>>(bufA, scsh + 128, scsh + 192, h1, N_ * HID);
    gin_pool_kernel<<<ewGrid, 256, 0, stream>>>(h1, batch, pooled + G_ * HID, N_ * HID);

    // ---- Layer 1 ----
    hipMemsetAsync(bufB, 0, nhB, stream);
    gin_scatter_kernel<<<scGrid, 256, 0, stream>>>(h1, src, dst, bufB, E_);
    gin_gemm_kernel<<<gemmGrid, 256, 0, stream>>>(h1, bufB, nullptr, nullptr,
        wf16 + 2 * 4096, c1_b1, bufA, stats + 256, stats + 320, 0);
    gin_bnfin_kernel<<<1, 64, 0, stream>>>(stats + 256, stats + 320, c1_g1, c1_be1,
        scsh + 256, scsh + 320, invN);
    gin_gemm_kernel<<<gemmGrid, 256, 0, stream>>>(bufA, nullptr, scsh + 256, scsh + 320,
        wf16 + 3 * 4096, c1_b2, bufB, stats + 384, stats + 448, 1);
    gin_bnfin_kernel<<<1, 64, 0, stream>>>(stats + 384, stats + 448, bn1_g, bn1_b,
        scsh + 384, scsh + 448, invN);
    gin_bnrelu_kernel<<<ewGrid, 256, 0, stream>>>(bufB, scsh + 384, scsh + 448, h2, N_ * HID);
    gin_pool_kernel<<<ewGrid, 256, 0, stream>>>(h2, batch, pooled + 2 * G_ * HID, N_ * HID);

    // ---- Readout ----
    gin_readout_kernel<<<(G_ * 10 + 255) / 256, 256, 0, stream>>>(
        pooled, pooled + G_ * HID, pooled + 2 * G_ * HID,
        l0_w, l0_b, l1_w, l1_b, l2_w, l2_b, out, G_);
}
